// GraphQNetwork_90443421319518
// MI455X (gfx1250) — matrix operand order repaired and checked by reading, not compiled, not run on hardware
//
#include <hip/hip_runtime.h>

typedef __attribute__((ext_vector_type(16))) __bf16 v16bf;
typedef __attribute__((ext_vector_type(8)))  float  v8f;

#define NODE_F 128   // IN_F == HID == 128

// ---------------------------------------------------------------- utilities
__global__ void zero_f32_kernel(float* __restrict__ p, long n) {
    long i = (long)blockIdx.x * blockDim.x + threadIdx.x;
    if (i < n) p[i] = 0.0f;
}

__global__ void cvt_bf16_kernel(const float* __restrict__ a, __bf16* __restrict__ o, int n) {
    int i = blockIdx.x * blockDim.x + threadIdx.x;
    if (i < n) o[i] = (__bf16)a[i];
}

__global__ void degree_kernel(const int* __restrict__ edst, float* __restrict__ deg, int E) {
    int e = blockIdx.x * blockDim.x + threadIdx.x;
    if (e < E)
        __hip_atomic_fetch_add(&deg[edst[e]], 1.0f, __ATOMIC_RELAXED, __HIP_MEMORY_SCOPE_AGENT);
}

__global__ void rdeg_kernel(const float* __restrict__ deg, float* __restrict__ rdeg, int n) {
    int i = blockIdx.x * blockDim.x + threadIdx.x;
    if (i < n) rdeg[i] = 1.0f / fmaxf(deg[i], 1.0f);
}

// --------------------------------------------------- edge scatter (L2-bound)
// One wave per edge: 32 lanes x float4 = 128 floats.  Coalesced 512B gather of
// feat[src], then 4 no-return global_atomic_add_f32 per lane into agg[dst].
__global__ void scatter_kernel(const float* __restrict__ feat,
                               const int* __restrict__ esrc,
                               const int* __restrict__ edst,
                               float* __restrict__ agg, int E) {
    long t = (long)blockIdx.x * blockDim.x + threadIdx.x;
    int e = (int)(t >> 5);
    int lane = (int)(t & 31);
    if (e >= E) return;
    int s = esrc[e];
    int d = edst[e];
    const float4 v = *(const float4*)(feat + (long)s * NODE_F + lane * 4);
    float* dst = agg + (long)d * NODE_F + lane * 4;
    __hip_atomic_fetch_add(dst + 0, v.x, __ATOMIC_RELAXED, __HIP_MEMORY_SCOPE_AGENT);
    __hip_atomic_fetch_add(dst + 1, v.y, __ATOMIC_RELAXED, __HIP_MEMORY_SCOPE_AGENT);
    __hip_atomic_fetch_add(dst + 2, v.z, __ATOMIC_RELAXED, __HIP_MEMORY_SCOPE_AGENT);
    __hip_atomic_fetch_add(dst + 3, v.w, __ATOMIC_RELAXED, __HIP_MEMORY_SCOPE_AGENT);
}

// ------------------------------------------------- fused SAGE layer (WMMA)
// out = relu( feat @ Wself^T + (agg * rdeg) @ Wneigh^T + bias )
// Block: 16 rows x 128 cols, 256 threads = 8 waves, wave w -> col tile w.
// A panels staged in LDS as bf16; K-loop of 4 -> 8 v_wmma_f32_16x16x32_bf16.
__global__ __launch_bounds__(256) void sage_gemm_kernel(
    const float*  __restrict__ feat,
    const float*  __restrict__ agg,
    const float*  __restrict__ rdeg,
    const __bf16* __restrict__ wself,    // [128 out][128 in] row-major bf16
    const __bf16* __restrict__ wneigh,   // [128 out][128 in] row-major bf16
    const float*  __restrict__ bias,     // [128]
    float*        __restrict__ out)      // [nrows][128]
{
    __shared__ __bf16 fa[16][NODE_F];
    __shared__ __bf16 ga[16][NODE_F];

    const int row0 = blockIdx.x * 16;
    const int tid  = threadIdx.x;

    // stage & convert both A-panels (f32 -> bf16), fuse 1/deg scaling
    for (int i = tid; i < 16 * NODE_F; i += 256) {
        int r = i >> 7, c = i & 127;
        long g = (long)(row0 + r) * NODE_F + c;
        fa[r][c] = (__bf16)feat[g];
        ga[r][c] = (__bf16)(agg[g] * rdeg[row0 + r]);
    }
    __syncthreads();

    const int wave  = tid >> 5;         // 0..7 -> output column tile
    const int lane  = tid & 31;
    const int ncol0 = wave * 16;
    const int n     = ncol0 + (lane & 15);
    const int bofs  = (lane < 16) ? 0 : 16;   // B: lanes 16-31 hold K+16
    const int m     = lane & 15;              // A: row within tile
    const int kh    = (lane >> 4) << 3;       // A: lanes 16-31 hold K+16 (via kh << 1)

    v8f c = {};
    #pragma unroll
    for (int kk = 0; kk < 4; ++kk) {
        const int kbase = kk * 32;

        // A fragments, contiguous K order matching B:
        // K = kbase + (lane<16?0:16) + e
        v16bf af, ag;
        #pragma unroll
        for (int e = 0; e < 16; ++e) {
            int k = kbase + (kh << 1) + e;
            af[e] = fa[m][k];
            ag[e] = ga[m][k];
        }

        // B fragments: 16 contiguous bf16 (32B aligned) per lane
        const v16bf bs = *(const v16bf*)(wself  + (long)n * NODE_F + kbase + bofs);
        const v16bf bn = *(const v16bf*)(wneigh + (long)n * NODE_F + kbase + bofs);
        if (kk < 3) {
            __builtin_prefetch(wself  + (long)n * NODE_F + kbase + 32 + bofs, 0, 3);
            __builtin_prefetch(wneigh + (long)n * NODE_F + kbase + 32 + bofs, 0, 3);
        }

        c = __builtin_amdgcn_wmma_f32_16x16x32_bf16(false, af, false, bs,
                                                    (short)0, c, false, false);
        c = __builtin_amdgcn_wmma_f32_16x16x32_bf16(false, ag, false, bn,
                                                    (short)0, c, false, false);
    }

    // epilogue: bias + relu; C/D layout: VGPR r -> M = r + (lane<16?0:8)
    const float bv   = bias[n];
    const int   mrow = (lane < 16) ? 0 : 8;
    #pragma unroll
    for (int r = 0; r < 8; ++r) {
        float v = c[r] + bv;
        v = v > 0.0f ? v : 0.0f;
        out[(long)(row0 + mrow + r) * NODE_F + n] = v;
    }
}

// ------------------------------------------------------------------- head
// out[0] = sum_f Wfc[f]*max_s h[states[s]][f] + Wfc[128+f]*h[action][f] + bfc
__global__ void head_kernel(const float* __restrict__ h,
                            const int*   __restrict__ states,
                            const int*   __restrict__ actions,
                            const float* __restrict__ Wfc,
                            const float* __restrict__ bfc,
                            float* __restrict__ out, int n_states) {
    const int f = threadIdx.x;                 // 128 threads
    float m = -3.402823466e38f;
    for (int s = 0; s < n_states; ++s)
        m = fmaxf(m, h[(long)states[s] * NODE_F + f]);
    const float av = h[(long)actions[0] * NODE_F + f];
    float v = Wfc[f] * m + Wfc[NODE_F + f] * av;

    __shared__ float red[NODE_F];
    red[f] = v;
    __syncthreads();
    for (int s = 64; s > 0; s >>= 1) {
        if (f < s) red[f] += red[f + s];
        __syncthreads();
    }
    if (f == 0) out[0] = red[0] + bfc[0];
}

// ------------------------------------------------------------------ launch
extern "C" void kernel_launch(void* const* d_in, const int* in_sizes, int n_in,
                              void* d_out, int out_size, void* d_ws, size_t ws_size,
                              hipStream_t stream) {
    const float* x      = (const float*)d_in[0];
    const int*   esrc   = (const int*)  d_in[1];
    const int*   edst   = (const int*)  d_in[2];
    const int*   states = (const int*)  d_in[3];
    const int*   action = (const int*)  d_in[4];
    const float* W1s    = (const float*)d_in[5];
    const float* W1n    = (const float*)d_in[6];
    const float* b1     = (const float*)d_in[7];
    const float* W2s    = (const float*)d_in[8];
    const float* W2n    = (const float*)d_in[9];
    const float* b2     = (const float*)d_in[10];
    const float* Wfc    = (const float*)d_in[11];
    const float* bfc    = (const float*)d_in[12];

    const int N = in_sizes[0] / NODE_F;   // 50000
    const int E = in_sizes[1];            // 800000
    const int S = in_sizes[3];            // 256

    // workspace carve-up (256B aligned)
    size_t off = 0;
    char* base = (char*)d_ws;
    auto carve = [&](size_t bytes) -> void* {
        void* p = base + off;
        off += (bytes + 255) & ~(size_t)255;
        return p;
    };
    float*  deg   = (float*) carve((size_t)N * 4);
    float*  rdeg  = (float*) carve((size_t)N * 4);
    float*  agg   = (float*) carve((size_t)N * NODE_F * 4);
    float*  h1    = (float*) carve((size_t)N * NODE_F * 4);
    float*  h2    = (float*) carve((size_t)N * NODE_F * 4);
    __bf16* wb1s  = (__bf16*)carve((size_t)NODE_F * NODE_F * 2);
    __bf16* wb1n  = (__bf16*)carve((size_t)NODE_F * NODE_F * 2);
    __bf16* wb2s  = (__bf16*)carve((size_t)NODE_F * NODE_F * 2);
    __bf16* wb2n  = (__bf16*)carve((size_t)NODE_F * NODE_F * 2);
    if (off > ws_size) return;

    const int W    = NODE_F * NODE_F;
    const long agN = (long)N * NODE_F;

    // weights -> bf16 (once per launch; tiny)
    cvt_bf16_kernel<<<(W + 255) / 256, 256, 0, stream>>>(W1s, wb1s, W);
    cvt_bf16_kernel<<<(W + 255) / 256, 256, 0, stream>>>(W1n, wb1n, W);
    cvt_bf16_kernel<<<(W + 255) / 256, 256, 0, stream>>>(W2s, wb2s, W);
    cvt_bf16_kernel<<<(W + 255) / 256, 256, 0, stream>>>(W2n, wb2n, W);

    // degrees (shared by both layers)
    zero_f32_kernel<<<(N + 255) / 256, 256, 0, stream>>>(deg, N);
    degree_kernel<<<(E + 255) / 256, 256, 0, stream>>>(edst, deg, E);
    rdeg_kernel<<<(N + 255) / 256, 256, 0, stream>>>(deg, rdeg, N);

    const long scatterT = (long)E * 32;
    const int  scatterB = (int)((scatterT + 255) / 256);
    const int  gemmB    = N / 16;   // 50000 / 16 = 3125

    // ---- layer 1
    zero_f32_kernel<<<(int)((agN + 255) / 256), 256, 0, stream>>>(agg, agN);
    scatter_kernel<<<scatterB, 256, 0, stream>>>(x, esrc, edst, agg, E);
    sage_gemm_kernel<<<gemmB, 256, 0, stream>>>(x, agg, rdeg, wb1s, wb1n, b1, h1);

    // ---- layer 2
    zero_f32_kernel<<<(int)((agN + 255) / 256), 256, 0, stream>>>(agg, agN);
    scatter_kernel<<<scatterB, 256, 0, stream>>>(h1, esrc, edst, agg, E);
    sage_gemm_kernel<<<gemmB, 256, 0, stream>>>(h1, agg, rdeg, wb2s, wb2n, b2, h2);

    // ---- head
    head_kernel<<<1, NODE_F, 0, stream>>>(h2, states, action, Wfc, bfc,
                                          (float*)d_out, S);
}
